// HGNNJobRecommender_13632226198148
// MI455X (gfx1250) — matrix operand order repaired and checked by reading, not compiled, not run on hardware
//
#include <hip/hip_runtime.h>
#include <math.h>

// ---------------------------------------------------------------------------
// Types
// ---------------------------------------------------------------------------
typedef __attribute__((ext_vector_type(16))) __bf16        v16bf;
typedef __attribute__((ext_vector_type(8)))  float         v8f;
typedef __attribute__((ext_vector_type(4)))  float         v4f;
typedef __attribute__((ext_vector_type(4)))  unsigned int  v4u;
typedef __attribute__((ext_vector_type(2)))  unsigned int  v2u;

#define NJOB  50000
#define NTECH 30000
#define NTOT  80000

// ---------------------------------------------------------------------------
// Device helpers
// ---------------------------------------------------------------------------
__device__ __forceinline__ unsigned short f2bf(float f) {
  unsigned int u = __float_as_uint(f);
  u += 0x7fffu + ((u >> 16) & 1u);           // round to nearest even
  return (unsigned short)(u >> 16);
}
__device__ __forceinline__ float gelu_exact(float x) {
  return 0.5f * x * (1.0f + erff(x * 0.7071067811865475f));
}
// monotonic uint encoding of float for atomicMax
__device__ __forceinline__ unsigned int ordered_enc(float f) {
  unsigned int u = __float_as_uint(f);
  return (u & 0x80000000u) ? ~u : (u | 0x80000000u);
}
__device__ __forceinline__ float ordered_dec(unsigned int k) {
  return (k & 0x80000000u) ? __uint_as_float(k & 0x7fffffffu) : __uint_as_float(~k);
}

union FragU { v4u q[2]; v16bf v; };
union AccU  { v8f v; float f[8]; };

// ---------------------------------------------------------------------------
// Generic bf16 WMMA GEMM:  C = epilogue( A(f32,MxK) @ W(f32,KxN) + bias )
//   block tile 128x64, 8 wave32 (4x2), each wave 32x32 via 4 wmma tiles
//   actA: 0=none 1=gelu (applied converting A to bf16)
//   epilogue: if skip != null: C = sig(*skip)*(acc+bias) + (1-sig)*res
//             else if actOut==1: leaky(acc+bias)
//   batched over grid.z with aBatch/wBatch/cBatch element strides
// ---------------------------------------------------------------------------
__global__ void __launch_bounds__(256)
gemm_bf16_wmma(const float* __restrict__ A, long long lda, long long aBatch,
               const float* __restrict__ W, long long ldw, long long wBatch,
               const float* __restrict__ bias,
               float* __restrict__ C, long long ldc, long long cBatch,
               const float* __restrict__ res, long long ldres,
               const float* __restrict__ skip,
               int M, int N, int K, int actA, int actOut)
{
  __shared__ __align__(16) unsigned short lA[128 * 32];  // [row][k] bf16
  __shared__ __align__(16) unsigned short lB[64 * 32];   // [col][k] bf16 (W transposed)

  int z = blockIdx.z;
  A += (long long)z * aBatch;
  W += (long long)z * wBatch;
  C += (long long)z * cBatch;

  const int m0 = blockIdx.y * 128;
  const int n0 = blockIdx.x * 64;
  const int tid = threadIdx.x;
  const int lane = tid & 31, wid = tid >> 5;
  const int wm = wid & 3, wn = wid >> 2;      // 4 m-waves x 2 n-waves
  const int hf = lane >> 4, l15 = lane & 15;

  const v8f vz = {0.f, 0.f, 0.f, 0.f, 0.f, 0.f, 0.f, 0.f};
  v8f a00 = vz, a01 = vz, a10 = vz, a11 = vz;

  const int nk = K >> 5;
  for (int kt = 0; kt < nk; ++kt) {
    // ---- stage A: 128x32 floats -> bf16 in LDS ----
#pragma unroll
    for (int i = 0; i < 4; ++i) {
      int idx = tid + i * 256;            // 1024 float4 groups
      int row = idx >> 3, seg = idx & 7;
      int rg = m0 + row;
      v4f av = {0.f, 0.f, 0.f, 0.f};
      if (rg < M) av = *(const v4f*)(A + (long long)rg * lda + (kt * 32 + seg * 4));
      if (actA == 1) {
        av.x = gelu_exact(av.x); av.y = gelu_exact(av.y);
        av.z = gelu_exact(av.z); av.w = gelu_exact(av.w);
      }
      v2u pk;
      pk.x = (unsigned)f2bf(av.x) | ((unsigned)f2bf(av.y) << 16);
      pk.y = (unsigned)f2bf(av.z) | ((unsigned)f2bf(av.w) << 16);
      *(v2u*)&lA[row * 32 + seg * 4] = pk;
    }
    // ---- stage B: W[k0..k0+32)[n0..n0+64) transposed to [n][k] ----
#pragma unroll
    for (int i = 0; i < 2; ++i) {
      int idx = tid + i * 256;            // 512 groups of 4 cols
      int kk = idx >> 4, seg = idx & 15;
      long long kg = (long long)(kt * 32 + kk);
#pragma unroll
      for (int c2 = 0; c2 < 4; ++c2) {
        int ng = n0 + seg * 4 + c2;
        unsigned short hv = 0;
        if (ng < N) hv = f2bf(W[kg * ldw + ng]);
        lB[(seg * 4 + c2) * 32 + kk] = hv;
      }
    }
    __syncthreads();

    // ---- fragments (ISA 7.12.2 16-bit layouts) ----
    FragU fa0, fa1, fb0, fb1;
    {
      int r0 = wm * 32 + l15, r1 = r0 + 16;
      fa0.q[0] = *(const v4u*)&lA[r0 * 32 + hf * 8];
      fa0.q[1] = *(const v4u*)&lA[r0 * 32 + 16 + hf * 8];
      fa1.q[0] = *(const v4u*)&lA[r1 * 32 + hf * 8];
      fa1.q[1] = *(const v4u*)&lA[r1 * 32 + 16 + hf * 8];
      int c0 = wn * 32 + l15, c1 = c0 + 16;
      fb0.q[0] = *(const v4u*)&lB[c0 * 32 + hf * 8];
      fb0.q[1] = *(const v4u*)&lB[c0 * 32 + 16 + hf * 8];
      fb1.q[0] = *(const v4u*)&lB[c1 * 32 + hf * 8];
      fb1.q[1] = *(const v4u*)&lB[c1 * 32 + 16 + hf * 8];
    }
    a00 = __builtin_amdgcn_wmma_f32_16x16x32_bf16(false, fa0.v, false, fb0.v, (short)0, a00, false, false);
    a01 = __builtin_amdgcn_wmma_f32_16x16x32_bf16(false, fa0.v, false, fb1.v, (short)0, a01, false, false);
    a10 = __builtin_amdgcn_wmma_f32_16x16x32_bf16(false, fa1.v, false, fb0.v, (short)0, a10, false, false);
    a11 = __builtin_amdgcn_wmma_f32_16x16x32_bf16(false, fa1.v, false, fb1.v, (short)0, a11, false, false);
    __syncthreads();
  }

  // ---- epilogue ----
  float sg = 0.f, tg = 0.f;
  const bool useSkip = (skip != nullptr);
  if (useSkip) { sg = 1.f / (1.f + expf(-skip[0])); tg = 1.f - sg; }

  v8f accs[4] = {a00, a01, a10, a11};
#pragma unroll
  for (int mi = 0; mi < 2; ++mi) {
#pragma unroll
    for (int ni = 0; ni < 2; ++ni) {
      AccU au; au.v = accs[mi * 2 + ni];
      int col = n0 + wn * 32 + ni * 16 + l15;
      if (col >= N) continue;
      float bv = bias ? bias[col] : 0.f;
#pragma unroll
      for (int r = 0; r < 8; ++r) {
        int rowg = m0 + wm * 32 + mi * 16 + hf * 8 + r;
        if (rowg >= M) continue;
        float v = au.f[r] + bv;
        if (useSkip)           v = sg * v + tg * res[(long long)rowg * ldres + col];
        else if (actOut == 1)  v = (v >= 0.f) ? v : 0.2f * v;
        C[(long long)rowg * ldc + col] = v;
      }
    }
  }
}

// ---------------------------------------------------------------------------
// Attention logits + running segment max (per edge, per head)
// ---------------------------------------------------------------------------
__global__ void att_logits_kernel(const int* __restrict__ edge, int E,
                                  const float* __restrict__ kqv, int qRowOff,
                                  const float* __restrict__ ke, int keRowOff,
                                  const float* __restrict__ prel,
                                  float* __restrict__ aBuf, int aBase,
                                  unsigned int* __restrict__ mEnc, int giOff)
{
  int idx = blockIdx.x * 256 + threadIdx.x;
  if (idx >= E * 4) return;
  int e = idx >> 2, h = idx & 3;
  int src = edge[e], dst = edge[E + e];
  const float* q = kqv + (long long)(qRowOff + dst) * 768 + 256 + h * 64;
  const float* k = ke + (long long)(keRowOff + src) * 256 + h * 64;
  float acc = 0.f;
#pragma unroll
  for (int d = 0; d < 64; d += 4) {
    v4f qa = *(const v4f*)(q + d);
    v4f ka = *(const v4f*)(k + d);
    acc += qa.x * ka.x + qa.y * ka.y + qa.z * ka.z + qa.w * ka.w;
  }
  float a = acc * prel[h] * 0.125f;  // / sqrt(64)
  aBuf[(long long)(aBase + e) * 4 + h] = a;
  atomicMax(&mEnc[(giOff + dst) * 4 + h], ordered_enc(a));
}

__global__ void att_expsum_kernel(const int* __restrict__ edge, int E,
                                  float* __restrict__ aBuf, int aBase,
                                  const unsigned int* __restrict__ mEnc,
                                  float* __restrict__ den, int giOff)
{
  int idx = blockIdx.x * 256 + threadIdx.x;
  if (idx >= E * 4) return;
  int e = idx >> 2, h = idx & 3;
  int dst = edge[E + e];
  int gi = giOff + dst;
  float m = ordered_dec(mEnc[gi * 4 + h]);
  if (!(m > -3.0e38f)) m = 0.f;  // non-finite -> 0 (matches reference)
  long long ai = (long long)(aBase + e) * 4 + h;
  float ex = expf(aBuf[ai] - m);
  aBuf[ai] = ex;
  atomicAdd(&den[gi * 4 + h], ex);
}

__global__ void att_scatter_kernel(const int* __restrict__ edge, int E,
                                   const float* __restrict__ aBuf, int aBase,
                                   const float* __restrict__ den,
                                   const float* __restrict__ ve, int veRowOff,
                                   float* __restrict__ agg, int giOff)
{
  int idx = blockIdx.x * 256 + threadIdx.x;  // E*256 threads
  if (idx >= E * 256) return;
  int e = idx >> 8, c = idx & 255, h = c >> 6;
  int src = edge[e], dst = edge[E + e];
  int gi = giOff + dst;
  float w = aBuf[(long long)(aBase + e) * 4 + h] / (den[gi * 4 + h] + 1e-16f);
  atomicAdd(&agg[(long long)gi * 256 + c],
            w * ve[(long long)(veRowOff + src) * 256 + c]);
}

// ---------------------------------------------------------------------------
// BatchNorm (biased var over rows) + leaky ReLU, per-channel block
// ---------------------------------------------------------------------------
__global__ void bn_leaky_kernel(const float* __restrict__ x, int rows,
                                const float* __restrict__ gamma,
                                const float* __restrict__ beta,
                                float* __restrict__ y)
{
  __shared__ float sh[512];
  int c = blockIdx.x;       // 256 channels
  int tid = threadIdx.x;
  float s = 0.f, s2 = 0.f;
  for (int r = tid; r < rows; r += 256) {
    float v = x[(long long)r * 256 + c];
    s += v; s2 += v * v;
  }
  sh[tid] = s; sh[256 + tid] = s2;
  __syncthreads();
  for (int o = 128; o > 0; o >>= 1) {
    if (tid < o) { sh[tid] += sh[tid + o]; sh[256 + tid] += sh[256 + tid + o]; }
    __syncthreads();
  }
  float mu = sh[0] / (float)rows;
  float var = sh[256] / (float)rows - mu * mu;
  float inv = rsqrtf(var + 1e-5f) * gamma[c];
  float bb = beta[c];
  for (int r = tid; r < rows; r += 256) {
    float v = (x[(long long)r * 256 + c] - mu) * inv + bb;
    y[(long long)r * 256 + c] = (v >= 0.f) ? v : 0.2f * v;
  }
}

// ---------------------------------------------------------------------------
// Aux final layer: sigmoid(q(128) . w + b), one thread per row
// ---------------------------------------------------------------------------
__global__ void aux_dot_kernel(const float* __restrict__ q, int rows,
                               const float* __restrict__ w,
                               const float* __restrict__ b,
                               float* __restrict__ outp)
{
  int r = blockIdx.x * 256 + threadIdx.x;
  if (r >= rows) return;
  float acc = b[0];
  const float* qr = q + (long long)r * 128;
#pragma unroll
  for (int j = 0; j < 128; j += 4) {
    v4f qa = *(const v4f*)(qr + j);
    v4f wa = *(const v4f*)(w + j);
    acc += qa.x * wa.x + qa.y * wa.y + qa.z * wa.z + qa.w * wa.w;
  }
  outp[r] = 1.f / (1.f + expf(-acc));
}

__global__ void fill_u32_kernel(unsigned int* __restrict__ p, long long n, unsigned int v)
{
  long long i = (long long)blockIdx.x * 256 + threadIdx.x;
  if (i < n) p[i] = v;
}

// ---------------------------------------------------------------------------
// Host launchers
// ---------------------------------------------------------------------------
static inline void gemm(hipStream_t st, const float* A, long long lda, long long aB,
                        const float* W, long long ldw, long long wB,
                        const float* bias, float* C, long long ldc, long long cB,
                        const float* res, long long ldres, const float* skip,
                        int M, int N, int K, int actA, int actOut, int Z)
{
  dim3 grid((unsigned)((N + 63) / 64), (unsigned)((M + 127) / 128), (unsigned)Z);
  gemm_bf16_wmma<<<grid, dim3(256), 0, st>>>(A, lda, aB, W, ldw, wB, bias,
                                             C, ldc, cB, res, ldres, skip,
                                             M, N, K, actA, actOut);
}
static inline void fill_u32(hipStream_t st, void* p, long long n, unsigned int v)
{
  fill_u32_kernel<<<dim3((unsigned)((n + 255) / 256)), dim3(256), 0, st>>>(
      (unsigned int*)p, n, v);
}

// ---------------------------------------------------------------------------
// kernel_launch
//   d_in: x_job, x_tech, edge_jt, edge_tj, edge_tt, then params flattened in
//   JAX pytree order (dicts sorted by key): aux(16) bn(4) cls(6) convs(2x13) proj(4)
// ---------------------------------------------------------------------------
extern "C" void kernel_launch(void* const* d_in, const int* in_sizes, int n_in,
                              void* d_out, int out_size, void* d_ws, size_t ws_size,
                              hipStream_t stream)
{
  (void)in_sizes; (void)n_in; (void)out_size;

  const float* x_job  = (const float*)d_in[0];
  const float* x_tech = (const float*)d_in[1];
  const int* eptr[3] = { (const int*)d_in[2], (const int*)d_in[3], (const int*)d_in[4] };
  auto P = [&](int i) { return (const float*)d_in[i]; };

  float* ws  = (float*)d_ws;
  float* out = (float*)d_out;

  // workspace layout (floats); agg and head scratch overlay the kqv region
  const long long W_HA = 0, W_HB = 20480000LL, W_KQV = 40960000LL,
                  W_KE = 102400000LL, W_VE = 130560000LL, W_A = 158720000LL,
                  W_M = 159920000LL, W_DEN = 160240000LL, W_TOT = 160560000LL;
  const long long W_Z1 = W_KQV, W_Z2 = 66560000LL, W_Q = 79360000LL;
  if (ws_size < (size_t)W_TOT * 4) return;  // insufficient scratch

  float* hA  = ws + W_HA;
  float* hB  = ws + W_HB;
  float* kqv = ws + W_KQV;
  float* agg = ws + W_KQV;   // overlays kqv (dead by scatter time)
  float* ke  = ws + W_KE;
  float* ve  = ws + W_VE;
  float* aB  = ws + W_A;
  unsigned int* mEnc = (unsigned int*)(ws + W_M);
  float* den = ws + W_DEN;

  // edge-type metadata: (job->tech), (tech->job), (tech->tech)
  const int Ecnt[3]        = {120000, 120000, 60000};
  const int aBase[3]       = {0, 120000, 240000};
  const long long kvRow[3] = {0, 50000, 80000};   // rows into ke/ve buffers
  const long long srcRow[3]= {0, 50000, 50000};   // source rows in kqv/h
  const int giOff[3]       = {50000, 0, 50000};   // OFF[dst type]

  // ---- input projections ----
  gemm(stream, x_job, 384, 0, P(58), 256, 0, P(57),
       hA, 256, 0, nullptr, 0, nullptr, NJOB, 256, 384, 0, 0, 1);
  gemm(stream, x_tech, 768, 0, P(60), 256, 0, P(59),
       hA + (long long)NJOB * 256, 256, 0, nullptr, 0, nullptr, NTECH, 256, 768, 0, 0, 1);

  for (int l = 0; l < 2; ++l) {
    int pb = 31 + l * 13;  // convs[l]: k_rel, kqvJ.b/.w, kqvT.b/.w, outJ.b/.w, outT.b/.w, p_rel, skipJ, skipT, v_rel
    fill_u32(stream, mEnc, (long long)NTOT * 4, 0x007FFFFFu);  // enc(-inf)
    fill_u32(stream, den,  (long long)NTOT * 4, 0u);

    // kqv projections
    gemm(stream, hA, 256, 0, P(pb + 2), 768, 0, P(pb + 1),
         kqv, 768, 0, nullptr, 0, nullptr, NJOB, 768, 256, 0, 0, 1);
    gemm(stream, hA + (long long)NJOB * 256, 256, 0, P(pb + 4), 768, 0, P(pb + 3),
         kqv + (long long)NJOB * 768, 768, 0, nullptr, 0, nullptr, NTECH, 768, 256, 0, 0, 1);

    // relational K/V transforms, batched over 4 heads via grid.z
    for (int e = 0; e < 3; ++e) {
      int Ms = (e == 0) ? NJOB : NTECH;
      gemm(stream, kqv + srcRow[e] * 768, 768, 64,           // A: k-part, head stride 64
           P(pb + 0) + e * 16384, 64, 4096, nullptr,         // k_rel[e,h] 64x64
           ke + kvRow[e] * 256, 256, 64,
           nullptr, 0, nullptr, Ms, 64, 64, 0, 0, 4);
      gemm(stream, kqv + srcRow[e] * 768 + 512, 768, 64,     // A: v-part
           P(pb + 12) + e * 16384, 64, 4096, nullptr,
           ve + kvRow[e] * 256, 256, 64,
           nullptr, 0, nullptr, Ms, 64, 64, 0, 0, 4);
    }

    // attention logits + segment max
    for (int e = 0; e < 3; ++e) {
      int nthr = Ecnt[e] * 4;
      att_logits_kernel<<<(nthr + 255) / 256, 256, 0, stream>>>(
          eptr[e], Ecnt[e], kqv, giOff[e], ke, (int)kvRow[e],
          P(pb + 9) + e * 4, aB, aBase[e], mEnc, giOff[e]);
    }
    // exp + segment sum
    for (int e = 0; e < 3; ++e) {
      int nthr = Ecnt[e] * 4;
      att_expsum_kernel<<<(nthr + 255) / 256, 256, 0, stream>>>(
          eptr[e], Ecnt[e], aB, aBase[e], mEnc, den, giOff[e]);
    }
    // weighted message scatter (kqv region is dead now -> agg overlays it)
    fill_u32(stream, agg, (long long)NTOT * 256, 0u);
    for (int e = 0; e < 3; ++e) {
      att_scatter_kernel<<<Ecnt[e], 256, 0, stream>>>(
          eptr[e], Ecnt[e], aB, aBase[e], den, ve, (int)kvRow[e], agg, giOff[e]);
    }

    // out = sig(skip)*lin(out_w, gelu(agg)) + (1-sig)*h
    gemm(stream, agg, 256, 0, P(pb + 6), 256, 0, P(pb + 5),
         hB, 256, 0, hA, 256, P(pb + 10), NJOB, 256, 256, 1, 0, 1);
    gemm(stream, agg + (long long)NJOB * 256, 256, 0, P(pb + 8), 256, 0, P(pb + 7),
         hB + (long long)NJOB * 256, 256, 0,
         hA + (long long)NJOB * 256, 256, P(pb + 11), NTECH, 256, 256, 1, 0, 1);

    // batch norm + leaky (final layer writes straight into output h regions)
    float* yJ = (l == 0) ? hA : (out + 1760000);
    float* yT = (l == 0) ? (hA + (long long)NJOB * 256) : (out + 14560000);
    bn_leaky_kernel<<<256, 256, 0, stream>>>(hB, NJOB, P(22), P(21), yJ);
    bn_leaky_kernel<<<256, 256, 0, stream>>>(hB + (long long)NJOB * 256, NTECH,
                                             P(24), P(23), yT);
  }

  const float* hJ = out + 1760000;
  const float* hT = out + 14560000;
  float* z1 = ws + W_Z1;
  float* z2 = ws + W_Z2;
  float* qb = ws + W_Q;

  // classifier: 256 -> 512 -> 256 -> 32
  gemm(stream, hJ, 256, 0, P(26), 512, 0, P(25),
       z1, 512, 0, nullptr, 0, nullptr, NJOB, 512, 256, 0, 1, 1);
  gemm(stream, z1, 512, 0, P(28), 256, 0, P(27),
       z2, 256, 0, nullptr, 0, nullptr, NJOB, 256, 512, 0, 1, 1);
  gemm(stream, z2, 256, 0, P(30), 32, 0, P(29),
       out, 32, 0, nullptr, 0, nullptr, NJOB, 32, 256, 0, 0, 1);

  // aux heads: (job,demand),(job,hotness),(tech,demand),(tech,hotness)
  struct Aux { const float* h; int rows; int b0, w0, b1, w1; long long oOff; };
  const Aux auxs[4] = {
    { hJ, NJOB,  5,  6,  7,  8, 1600000LL },
    { hJ, NJOB,  9, 10, 11, 12, 1650000LL },
    { hT, NTECH, 13, 14, 15, 16, 1700000LL },
    { hT, NTECH, 17, 18, 19, 20, 1730000LL },
  };
  for (int i = 0; i < 4; ++i) {
    gemm(stream, auxs[i].h, 256, 0, P(auxs[i].w0), 128, 0, P(auxs[i].b0),
         qb, 128, 0, nullptr, 0, nullptr, auxs[i].rows, 128, 256, 0, 1, 1);
    aux_dot_kernel<<<(auxs[i].rows + 255) / 256, 256, 0, stream>>>(
        qb, auxs[i].rows, P(auxs[i].w1), P(auxs[i].b1), out + auxs[i].oOff);
  }
}
